// MaskDecoderLayer_85366769975695
// MI455X (gfx1250) — matrix operand order repaired and checked by reading, not compiled, not run on hardware
//
#include <hip/hip_runtime.h>
#include <math.h>

// ---------------------------------------------------------------------------
// MaskDecoderLayer for MI455X (gfx1250): all matmuls via v_wmma_f32_16x16x32_bf16
// v2: branchless clamped staging, software-pipelined global->LDS (regs in
// flight across the WMMA phase), global_prefetch, compile-time epilogue divisor.
// ---------------------------------------------------------------------------

typedef __bf16 bf16;
typedef __attribute__((ext_vector_type(16))) __bf16 v16bf;
typedef __attribute__((ext_vector_type(8)))  float  v8f;
typedef unsigned int u32x4 __attribute__((ext_vector_type(4)));

union FragBF { v16bf v; u32x4 q[2]; };

#define DEV __device__ __forceinline__

DEV bf16 f2bf(float f) {
    unsigned u = __builtin_bit_cast(unsigned, f);
    unsigned r = (u + 0x7FFFu + ((u >> 16) & 1u)) >> 16;
    unsigned short s = (unsigned short)r;
    return __builtin_bit_cast(bf16, s);
}
DEV float bf2f(bf16 b) {
    unsigned short s = __builtin_bit_cast(unsigned short, b);
    unsigned u = ((unsigned)s) << 16;
    return __builtin_bit_cast(float, u);
}
DEV v8f wmma_bf16(const v16bf& a, const v16bf& b, const v8f& c) {
    return __builtin_amdgcn_wmma_f32_16x16x32_bf16(false, a, false, b, (short)0, c, false, false);
}
DEV int imin(int a, int b) { return a < b ? a : b; }

// ---------------------------------------------------------------------------
// fp32 -> bf16 convert (grid-stride)
// ---------------------------------------------------------------------------
__global__ void cvt_bf16_kernel(const float* __restrict__ in, bf16* __restrict__ out, long long n) {
    long long i = (long long)blockIdx.x * blockDim.x + threadIdx.x;
    long long stride = (long long)gridDim.x * blockDim.x;
    for (; i < n; i += stride) out[i] = f2bf(in[i]);
}

// ---------------------------------------------------------------------------
// bias precompute: region = prev_mask > 0 (sigmoid(x)>0.5); if row has no
// positive -> all zeros bias; else bias = region ? 0 : -10000.
// One block per (b,q) row of 4096.
// ---------------------------------------------------------------------------
__global__ __launch_bounds__(256) void bias_kernel(const float* __restrict__ pm, bf16* __restrict__ out) {
    int row = blockIdx.x, tid = threadIdx.x;
    const float* p = pm + (size_t)row * 4096;
    float x[16]; int any = 0;
#pragma unroll
    for (int i = 0; i < 16; i++) { x[i] = p[tid + i * 256]; any |= (x[i] > 0.f) ? 1 : 0; }
    __shared__ int sh;
    if (tid == 0) sh = 0;
    __syncthreads();
    unsigned long long bl = __ballot(any);
    if ((tid & 31) == 0 && bl) atomicOr(&sh, 1);
    __syncthreads();
    int rowAny = sh;
    bf16* o = out + (size_t)row * 4096;
#pragma unroll
    for (int i = 0; i < 16; i++) {
        float bias = rowAny ? ((x[i] > 0.f) ? 0.f : -10000.f) : 0.f;
        o[tid + i * 256] = f2bf(bias);
    }
}

// ---------------------------------------------------------------------------
// LayerNorm over D=256, one row per 256-thread block, bf16 output
// ---------------------------------------------------------------------------
__global__ __launch_bounds__(256) void ln_kernel(const float* __restrict__ x, const float* __restrict__ g,
                                                 const float* __restrict__ b, bf16* __restrict__ out) {
    int row = blockIdx.x, tid = threadIdx.x;
    float v = x[(size_t)row * 256 + tid];
    __shared__ float red[8];
    float s = v;
#pragma unroll
    for (int o = 16; o; o >>= 1) s += __shfl_xor(s, o, 32);
    if ((tid & 31) == 0) red[tid >> 5] = s;
    __syncthreads();
    float mean = 0.f;
#pragma unroll
    for (int i = 0; i < 8; i++) mean += red[i];
    mean *= (1.f / 256.f);
    __syncthreads();
    float d = v - mean;
    s = d * d;
#pragma unroll
    for (int o = 16; o; o >>= 1) s += __shfl_xor(s, o, 32);
    if ((tid & 31) == 0) red[tid >> 5] = s;
    __syncthreads();
    float var = 0.f;
#pragma unroll
    for (int i = 0; i < 8; i++) var += red[i];
    var *= (1.f / 256.f);
    out[(size_t)row * 256 + tid] = f2bf(d * rsqrtf(var + 1e-5f) * g[tid] + b[tid]);
}

// ---------------------------------------------------------------------------
// Generic WMMA GEMM: C[M,N] = A_bf16[M,K] @ W_bf16[N,K]^T + bias[N]  (+ epilogue)
// 64x64 tile, 128 threads (4 waves), each wave owns 16 M-rows x 64 N-cols.
// Staging is branchless (row index clamped to M-1; clamped rows are never
// stored) and software-pipelined: next K-tile rides in registers while the
// current LDS tile feeds WMMAs.
// ---------------------------------------------------------------------------
enum { EPI_QKV = 0, EPI_RESID = 1, EPI_GELU = 2 };

struct GemmP {
    const bf16* A; const bf16* W; const float* bias; const float* resid;
    float* outF; bf16* outB; bf16* outK; bf16* outV;
    int M, N, K, nq, Slen, SlenPad, ldo;
};

template <int EPI, int ROWS>
__global__ __launch_bounds__(128) void gemm_wmma(GemmP p) {
    __shared__ bf16 As[64 * 40];   // 64 rows x 32 K, padded to 40 (80B rows, conflict-free b128)
    __shared__ bf16 Bs[64 * 40];   // 64 N-rows x 32 K
    const int tid = threadIdx.x, lane = tid & 31, wv = tid >> 5, half = lane >> 4, l16 = lane & 15;
    const int n0 = blockIdx.x * 64, m0 = blockIdx.y * 64;

    // staging coordinates: each thread owns 2 of the 256 16B chunks per tile
    const int c0 = tid, c1 = tid + 128;
    const int ar0 = c0 >> 2, ak0 = (c0 & 3) * 8;
    const int ar1 = c1 >> 2, ak1 = (c1 & 3) * 8;
    const size_t amo0 = (size_t)imin(m0 + ar0, p.M - 1) * p.K;
    const size_t amo1 = (size_t)imin(m0 + ar1, p.M - 1) * p.K;
    const size_t bno0 = (size_t)(n0 + ar0) * p.K;   // N is always a multiple of 64
    const size_t bno1 = (size_t)(n0 + ar1) * p.K;

    u32x4 ra0 = *(const u32x4*)(p.A + amo0 + ak0);
    u32x4 ra1 = *(const u32x4*)(p.A + amo1 + ak1);
    u32x4 rb0 = *(const u32x4*)(p.W + bno0 + ak0);
    u32x4 rb1 = *(const u32x4*)(p.W + bno1 + ak1);

    v8f acc[4];
#pragma unroll
    for (int f = 0; f < 4; f++) acc[f] = (v8f){0.f,0.f,0.f,0.f,0.f,0.f,0.f,0.f};

    for (int k0 = 0; k0 < p.K; k0 += 32) {
        *(u32x4*)(&As[ar0 * 40 + ak0]) = ra0;
        *(u32x4*)(&As[ar1 * 40 + ak1]) = ra1;
        *(u32x4*)(&Bs[ar0 * 40 + ak0]) = rb0;
        *(u32x4*)(&Bs[ar1 * 40 + ak1]) = rb1;
        __syncthreads();
        const int kn = k0 + 32;
        if (kn < p.K) {                      // uniform: prefetch next tile into regs
            ra0 = *(const u32x4*)(p.A + amo0 + kn + ak0);
            ra1 = *(const u32x4*)(p.A + amo1 + kn + ak1);
            rb0 = *(const u32x4*)(p.W + bno0 + kn + ak0);
            rb1 = *(const u32x4*)(p.W + bno1 + kn + ak1);
            if (kn + 32 < p.K) {             // L2 prefetch two tiles ahead
                __builtin_prefetch(p.A + amo0 + kn + 32 + ak0, 0, 3);
                __builtin_prefetch(p.W + bno0 + kn + 32 + ak0, 0, 3);
            }
        }
        FragBF a;                            // A frag: 16x32 (M=l16, K per ISA layout)
        a.q[0] = *(const u32x4*)(&As[(wv * 16 + l16) * 40 + half * 8]);
        a.q[1] = *(const u32x4*)(&As[(wv * 16 + l16) * 40 + 16 + half * 8]);
#pragma unroll
        for (int f = 0; f < 4; f++) {        // B frags: 32x16 (N=l16, K per ISA layout)
            FragBF bb;
            bb.q[0] = *(const u32x4*)(&Bs[(f * 16 + l16) * 40 + half * 8]);
            bb.q[1] = *(const u32x4*)(&Bs[(f * 16 + l16) * 40 + 16 + half * 8]);
            acc[f] = wmma_bf16(a.v, bb.v, acc[f]);
        }
        __syncthreads();
    }

#pragma unroll
    for (int f = 0; f < 4; f++) {
        int n = n0 + f * 16 + l16;
#pragma unroll
        for (int r = 0; r < 8; r++) {
            int m = m0 + wv * 16 + r + 8 * half;           // C layout: M = r + 8*half
            if (m >= p.M) continue;
            float v = acc[f][r] + p.bias[n];
            if constexpr (EPI == EPI_RESID) {
                p.outF[(size_t)m * p.ldo + n] = v + p.resid[(size_t)m * p.ldo + n];
            } else if constexpr (EPI == EPI_GELU) {
                float gl = 0.5f * v * (1.f + erff(v * 0.70710678118f));
                p.outB[(size_t)m * p.ldo + n] = f2bf(gl);
            } else { // EPI_QKV: scatter into head layouts; V stored transposed (b,h,hd,s)
                int b = m / ROWS, row = m - b * ROWS;      // ROWS compile-time -> mul/shift
                int nn = n;
                if (nn < p.nq) {
                    int hh = nn >> 5, hd = nn & 31;
                    p.outB[(((size_t)(b * 8 + hh)) * 100 + row) * 32 + hd] = f2bf(v);
                } else {
                    nn -= p.nq;
                    if (nn < 256) {
                        int hh = nn >> 5, hd = nn & 31;
                        p.outK[(((size_t)(b * 8 + hh)) * p.Slen + row) * 32 + hd] = f2bf(v);
                    } else {
                        nn -= 256;
                        int hh = nn >> 5, hd = nn & 31;
                        p.outV[(((size_t)(b * 8 + hh)) * 32 + hd) * p.SlenPad + row] = f2bf(v);
                    }
                }
            }
        }
    }
}

// ---------------------------------------------------------------------------
// Flash attention, one workgroup per (b,h); 8 waves x 16 q-rows = 128 (Q=100
// padded), S chunks of 64. Online softmax via half-wave shfl reductions.
// K/V chunk staging is branchless (clamped; padded s columns are masked to
// -30000 pre-softmax) and software-pipelined like the GEMM.
// ---------------------------------------------------------------------------
template <bool USE_BIAS>
__global__ __launch_bounds__(256) void attn_kernel(const bf16* __restrict__ qh, const bf16* __restrict__ kh,
                                                   const bf16* __restrict__ vh, const bf16* __restrict__ biasAll,
                                                   bf16* __restrict__ out, int Qn, int Slen, int SlenPad,
                                                   float scale) {
    __shared__ bf16 qs[128 * 40];      // q tile [128][32] pad 40
    __shared__ bf16 ks[64 * 40];       // k chunk [64 s][32 hd]
    __shared__ bf16 vs[32 * 72];       // v chunk [32 hd][64 s] pad 72
    __shared__ bf16 ps[8 * 16 * 72];   // per-wave probs scratch [16][64] pad 72

    const int tid = threadIdx.x, lane = tid & 31, wv = tid >> 5, half = lane >> 4, l16 = lane & 15;
    const int bh = blockIdx.x, b = bh >> 3, h = bh & 7;
    const bf16* qg = qh + (size_t)bh * Qn * 32;
    const bf16* kg = kh + (size_t)bh * Slen * 32;
    const bf16* vg = vh + (size_t)bh * 32 * SlenPad;
    const bf16* biasB = USE_BIAS ? (biasAll + (size_t)b * Qn * Slen) : nullptr;

    // staging coords
    const int krow = tid >> 2, kkc = (tid & 3) * 8;           // k chunk: 64x32
    const int vrow = tid >> 3, vsc = (tid & 7) * 8;           // v chunk: 32x64

    for (int c = tid; c < 512; c += 256) {   // stage q once (clamped pad rows; never stored)
        int row = c >> 2, kc = (c & 3) * 8;
        int rsrc = imin(row, Qn - 1);
        *(u32x4*)(&qs[row * 40 + kc]) = *(const u32x4*)(qg + (size_t)rsrc * 32 + kc);
    }
    __syncthreads();
    FragBF qa;
    qa.q[0] = *(const u32x4*)(&qs[(wv * 16 + l16) * 40 + half * 8]);
    qa.q[1] = *(const u32x4*)(&qs[(wv * 16 + l16) * 40 + 16 + half * 8]);

    float mrow[8], lrow[8];
#pragma unroll
    for (int r = 0; r < 8; r++) { mrow[r] = -3.0e38f; lrow[r] = 0.f; }
    v8f o0 = (v8f){0.f,0.f,0.f,0.f,0.f,0.f,0.f,0.f};
    v8f o1 = o0;
    const int qrow_base = wv * 16 + 8 * half;

    // pipelined staging registers for chunk 0
    u32x4 kreg = *(const u32x4*)(kg + (size_t)imin(krow, Slen - 1) * 32 + kkc);
    u32x4 vreg = *(const u32x4*)(vg + (size_t)vrow * SlenPad + vsc);

    for (int s0 = 0; s0 < SlenPad; s0 += 64) {
        *(u32x4*)(&ks[krow * 40 + kkc]) = kreg;
        *(u32x4*)(&vs[vrow * 72 + vsc]) = vreg;
        __syncthreads();
        const int sn = s0 + 64;
        if (sn < SlenPad) {                   // uniform: next chunk into regs
            kreg = *(const u32x4*)(kg + (size_t)imin(sn + krow, Slen - 1) * 32 + kkc);
            vreg = *(const u32x4*)(vg + (size_t)vrow * SlenPad + sn + vsc);
        }

        v8f sc4[4];
#pragma unroll
        for (int f = 0; f < 4; f++) {         // QK^T : 4 WMMAs -> 16x64 scores
            FragBF kb;
            kb.q[0] = *(const u32x4*)(&ks[(f * 16 + l16) * 40 + half * 8]);
            kb.q[1] = *(const u32x4*)(&ks[(f * 16 + l16) * 40 + 16 + half * 8]);
            v8f z = (v8f){0.f,0.f,0.f,0.f,0.f,0.f,0.f,0.f};
            sc4[f] = wmma_bf16(qa.v, kb.v, z);
        }
#pragma unroll
        for (int f = 0; f < 4; f++) {         // scale + bias + padding mask
            int s = s0 + f * 16 + l16;
#pragma unroll
            for (int r = 0; r < 8; r++) {
                float val = sc4[f][r] * scale;
                if (USE_BIAS) {
                    int qr = qrow_base + r; int qc = (qr < Qn) ? qr : 0;
                    int ss = (s < Slen) ? s : 0;
                    val += bf2f(biasB[(size_t)qc * Slen + ss]);
                }
                if (s >= Slen) val = -30000.f;
                sc4[f][r] = val;
            }
        }
#pragma unroll
        for (int r = 0; r < 8; r++) {         // online softmax row update
            float vmax = fmaxf(fmaxf(sc4[0][r], sc4[1][r]), fmaxf(sc4[2][r], sc4[3][r]));
            vmax = fmaxf(vmax, __shfl_xor(vmax, 1, 32));
            vmax = fmaxf(vmax, __shfl_xor(vmax, 2, 32));
            vmax = fmaxf(vmax, __shfl_xor(vmax, 4, 32));
            vmax = fmaxf(vmax, __shfl_xor(vmax, 8, 32));
            float mnew = fmaxf(mrow[r], vmax);
            float corr = __expf(mrow[r] - mnew);
            float lsum = 0.f;
#pragma unroll
            for (int f = 0; f < 4; f++) { float pp = __expf(sc4[f][r] - mnew); sc4[f][r] = pp; lsum += pp; }
            lsum += __shfl_xor(lsum, 1, 32);
            lsum += __shfl_xor(lsum, 2, 32);
            lsum += __shfl_xor(lsum, 4, 32);
            lsum += __shfl_xor(lsum, 8, 32);
            lrow[r] = lrow[r] * corr + lsum;
            mrow[r] = mnew;
            o0[r] *= corr; o1[r] *= corr;
        }
        bf16* psw = ps + wv * 16 * 72;        // C-layout -> A-layout via wave-private LDS
#pragma unroll
        for (int f = 0; f < 4; f++)
#pragma unroll
            for (int r = 0; r < 8; r++)
                psw[(r + 8 * half) * 72 + f * 16 + l16] = f2bf(sc4[f][r]);
#pragma unroll
        for (int kb2 = 0; kb2 < 2; kb2++) {   // P@V : 4 WMMAs (K=64, N=32)
            FragBF pa, vb0, vb1;
            pa.q[0] = *(const u32x4*)(&psw[l16 * 72 + kb2 * 32 + half * 8]);
            pa.q[1] = *(const u32x4*)(&psw[l16 * 72 + kb2 * 32 + 16 + half * 8]);
            vb0.q[0] = *(const u32x4*)(&vs[l16 * 72 + kb2 * 32 + half * 8]);
            vb0.q[1] = *(const u32x4*)(&vs[l16 * 72 + kb2 * 32 + 16 + half * 8]);
            vb1.q[0] = *(const u32x4*)(&vs[(16 + l16) * 72 + kb2 * 32 + half * 8]);
            vb1.q[1] = *(const u32x4*)(&vs[(16 + l16) * 72 + kb2 * 32 + 16 + half * 8]);
            o0 = wmma_bf16(pa.v, vb0.v, o0);
            o1 = wmma_bf16(pa.v, vb1.v, o1);
        }
        __syncthreads();
    }

#pragma unroll
    for (int r = 0; r < 8; r++) {             // out (b,q,h,hd) bf16 row-major [800][256]
        int qr = qrow_base + r;
        if (qr >= Qn) continue;
        float inv = 1.f / lrow[r];
        size_t base = ((size_t)(b * Qn + qr)) * 256 + h * 32;
        out[base + l16]      = f2bf(o0[r] * inv);
        out[base + 16 + l16] = f2bf(o1[r] * inv);
    }
}

// ---------------------------------------------------------------------------
// Host launcher. B=8,Q=100,S=4096,D=256,H=8,HD=32. Workspace ~64MB.
// ---------------------------------------------------------------------------
extern "C" void kernel_launch(void* const* d_in, const int* in_sizes, int n_in,
                              void* d_out, int out_size, void* d_ws, size_t ws_size,
                              hipStream_t stream) {
    (void)in_sizes; (void)n_in; (void)out_size; (void)ws_size;
    const float* queries     = (const float*)d_in[0];
    const float* pixel       = (const float*)d_in[1];
    const float* prev_mask   = (const float*)d_in[2];
    const float* cross_in_w  = (const float*)d_in[3];
    const float* cross_in_b  = (const float*)d_in[4];
    const float* cross_out_w = (const float*)d_in[5];
    const float* cross_out_b = (const float*)d_in[6];
    const float* self_in_w   = (const float*)d_in[7];
    const float* self_in_b   = (const float*)d_in[8];
    const float* self_out_w  = (const float*)d_in[9];
    const float* self_out_b  = (const float*)d_in[10];
    const float* ln_cross_g  = (const float*)d_in[11];
    const float* ln_cross_b  = (const float*)d_in[12];
    const float* ln_self_g   = (const float*)d_in[13];
    const float* ln_self_b   = (const float*)d_in[14];
    const float* ln_ffn_g    = (const float*)d_in[15];
    const float* ln_ffn_b    = (const float*)d_in[16];
    const float* ffn_w1      = (const float*)d_in[17];
    const float* ffn_b1      = (const float*)d_in[18];
    const float* ffn_w2      = (const float*)d_in[19];
    const float* ffn_b2      = (const float*)d_in[20];

    char* wp = (char*)d_ws;
    auto alloc = [&](size_t bytes) -> char* { char* r = wp; wp += (bytes + 255) & ~(size_t)255; return r; };

    bf16* pixel_bf = (bf16*)alloc(8ull * 4096 * 256 * 2);
    bf16* w_ci = (bf16*)alloc(768 * 256 * 2);
    bf16* w_co = (bf16*)alloc(256 * 256 * 2);
    bf16* w_si = (bf16*)alloc(768 * 256 * 2);
    bf16* w_so = (bf16*)alloc(256 * 256 * 2);
    bf16* w_f1 = (bf16*)alloc(1024 * 256 * 2);
    bf16* w_f2 = (bf16*)alloc(256 * 1024 * 2);
    bf16* biasb = (bf16*)alloc(800ull * 4096 * 2);
    bf16* lnq  = (bf16*)alloc(800 * 256 * 2);
    bf16* qh1  = (bf16*)alloc(64 * 100 * 32 * 2);
    bf16* k1   = (bf16*)alloc(64ull * 4096 * 32 * 2);
    bf16* v1   = (bf16*)alloc(64ull * 32 * 4096 * 2);
    bf16* at1  = (bf16*)alloc(800 * 256 * 2);
    float* q1  = (float*)alloc(800 * 256 * 4);
    bf16* ln2  = (bf16*)alloc(800 * 256 * 2);
    bf16* qh2  = (bf16*)alloc(64 * 100 * 32 * 2);
    bf16* k2   = (bf16*)alloc(64 * 100 * 32 * 2);
    bf16* v2   = (bf16*)alloc(64 * 32 * 128 * 2);
    bf16* at2  = (bf16*)alloc(800 * 256 * 2);
    float* q2  = (float*)alloc(800 * 256 * 4);
    bf16* ln3  = (bf16*)alloc(800 * 256 * 2);
    bf16* fh   = (bf16*)alloc(800ull * 1024 * 2);

    const float scale = 0.17677669529663689f; // 1/sqrt(32)

    // --- precision converts + bias precompute ---
    cvt_bf16_kernel<<<2048, 256, 0, stream>>>(pixel, pixel_bf, 8ll * 4096 * 256);
    cvt_bf16_kernel<<<512, 256, 0, stream>>>(cross_in_w, w_ci, 768 * 256);
    cvt_bf16_kernel<<<256, 256, 0, stream>>>(cross_out_w, w_co, 256 * 256);
    cvt_bf16_kernel<<<512, 256, 0, stream>>>(self_in_w, w_si, 768 * 256);
    cvt_bf16_kernel<<<256, 256, 0, stream>>>(self_out_w, w_so, 256 * 256);
    cvt_bf16_kernel<<<512, 256, 0, stream>>>(ffn_w1, w_f1, 1024 * 256);
    cvt_bf16_kernel<<<512, 256, 0, stream>>>(ffn_w2, w_f2, 256 * 1024);
    bias_kernel<<<800, 256, 0, stream>>>(prev_mask, biasb);

    // --- cross attention block ---
    ln_kernel<<<800, 256, 0, stream>>>(queries, ln_cross_g, ln_cross_b, lnq);
    GemmP gp{};
    // q projection -> (b,h,q,hd)
    gp = {lnq, w_ci, cross_in_b, nullptr, nullptr, qh1, nullptr, nullptr, 800, 256, 256, 256, 0, 0, 0};
    gemm_wmma<EPI_QKV, 100><<<dim3(4, 13), 128, 0, stream>>>(gp);
    // kv projection of pixel_feat -> K (b,h,s,hd), V^T (b,h,hd,s)
    gp = {pixel_bf, w_ci + 256 * 256, cross_in_b + 256, nullptr, nullptr, nullptr, k1, v1,
          32768, 512, 256, 0, 4096, 4096, 0};
    gemm_wmma<EPI_QKV, 4096><<<dim3(8, 512), 128, 0, stream>>>(gp);
    attn_kernel<true><<<64, 256, 0, stream>>>(qh1, k1, v1, biasb, at1, 100, 4096, 4096, scale);
    // out projection + residual -> q1 (fp32)
    gp = {at1, w_co, cross_out_b, queries, q1, nullptr, nullptr, nullptr, 800, 256, 256, 0, 0, 0, 256};
    gemm_wmma<EPI_RESID, 1><<<dim3(4, 13), 128, 0, stream>>>(gp);

    // --- self attention block ---
    ln_kernel<<<800, 256, 0, stream>>>(q1, ln_self_g, ln_self_b, ln2);
    gp = {ln2, w_si, self_in_b, nullptr, nullptr, qh2, k2, v2, 800, 768, 256, 256, 100, 128, 0};
    gemm_wmma<EPI_QKV, 100><<<dim3(12, 13), 128, 0, stream>>>(gp);
    attn_kernel<false><<<64, 256, 0, stream>>>(qh2, k2, v2, nullptr, at2, 100, 100, 128, scale);
    gp = {at2, w_so, self_out_b, q1, q2, nullptr, nullptr, nullptr, 800, 256, 256, 0, 0, 0, 256};
    gemm_wmma<EPI_RESID, 1><<<dim3(4, 13), 128, 0, stream>>>(gp);

    // --- FFN block ---
    ln_kernel<<<800, 256, 0, stream>>>(q2, ln_ffn_g, ln_ffn_b, ln3);
    gp = {ln3, w_f1, ffn_b1, nullptr, nullptr, fh, nullptr, nullptr, 800, 1024, 256, 0, 0, 0, 1024};
    gemm_wmma<EPI_GELU, 1><<<dim3(16, 13), 128, 0, stream>>>(gp);
    gp = {fh, w_f2, ffn_b2, q2, (float*)d_out, nullptr, nullptr, nullptr, 800, 256, 1024, 0, 0, 0, 256};
    gemm_wmma<EPI_RESID, 1><<<dim3(4, 13), 128, 0, stream>>>(gp);
}
